// LAModule_53901839565677
// MI455X (gfx1250) — matrix operand order repaired and checked by reading, not compiled, not run on hardware
//
#include <hip/hip_runtime.h>
#include <hip/hip_bf16.h>

typedef __attribute__((ext_vector_type(16))) _Float16 v16h;
typedef __attribute__((ext_vector_type(8)))  _Float16 v8h;
typedef __attribute__((ext_vector_type(4)))  _Float16 v4h;
typedef __attribute__((ext_vector_type(8)))  float    v8f;
typedef __attribute__((ext_vector_type(4)))  unsigned int u32x4;
typedef __attribute__((ext_vector_type(4)))  int      i32x4;
typedef __attribute__((ext_vector_type(8)))  int      i32x8;

#define DIMC   512
#define NTOK   8192
#define BATCH  4
#define NHEADS 8
#define HD     64
#define WIN    128
#define NWIN   (NTOK / WIN)          // 64
#define HEADSZ ((size_t)BATCH * NHEADS * NTOK * HD)   // 16,777,216 elements

__device__ __forceinline__ v8f vzero8f() {
    v8f z;
#pragma unroll
    for (int i = 0; i < 8; ++i) z[i] = 0.0f;
    return z;
}

// Build a 16-half A/B fragment from two contiguous 8-half chunks in LDS.
__device__ __forceinline__ v16h ldpair(const _Float16* p0, const _Float16* p1) {
    const v8h lo = *(const v8h*)p0;
    const v8h hi = *(const v8h*)p1;
    v16h r;
#pragma unroll
    for (int i = 0; i < 8; ++i) { r[i] = lo[i]; r[i + 8] = hi[i]; }
    return r;
}

__device__ __forceinline__ v8f wmma_f16(v16h a, v16h b, v8f c) {
    return __builtin_amdgcn_wmma_f32_16x16x32_f16(false, a, false, b, (short)0, c, false, false);
}

// ---------------------------------------------------------------------------
// TDM: 1-D fp16 tile, global -> LDS, inserting 16B of LDS padding after every
// 128B streamed (matches the [rows][72] half layout, row stride 144B).
//   pad_interval code 4 -> pad after 16 x 8B = 128B; pad_amount code 3 -> 4 DW.
// Issued once per wave (EXEC is ignored by TDM); tracked by TENSORcnt.
// This toolchain uses the 6-arg builtin form.
// ---------------------------------------------------------------------------
__device__ __forceinline__ void tdm_load_1d_f16(unsigned lds_byte_off,
                                                const _Float16* gptr,
                                                unsigned nelem) {
    const unsigned long long ga = (unsigned long long)(uintptr_t)gptr;
    u32x4 g0;
    g0[0] = 1u;                                   // count=1 (valid user D#)
    g0[1] = lds_byte_off;                         // lds_addr
    g0[2] = (unsigned)(ga & 0xffffffffu);         // global_addr[31:0]
    g0[3] = (unsigned)((ga >> 32) & 0x01ffffffu)  // global_addr[56:32]
          | (2u << 30);                           // type = 2 ("image")
    i32x8 g1;
    g1[0] = (1 << 16)                             // data_size = 2 bytes
          | (1 << 20)                             // pad_enable
          | (4 << 22)                             // pad_interval: every 128B
          | (3 << 25);                            // pad_amount: 4 DWORDs (16B)
    g1[1] = (int)((nelem & 0xffffu) << 16);       // tensor_dim0[15:0]
    g1[2] = (int)(((nelem >> 16) & 0xffffu)       // tensor_dim0[31:16]
          | (1u << 16));                          // tensor_dim1 = 1
    g1[3] = (int)((nelem & 0xffffu) << 16);       // tile_dim0 (16 bits)
    g1[4] = 1;                                    // tile_dim1 = 1
    g1[5] = (int)nelem;                           // tensor_dim0_stride[31:0]
    g1[6] = 0;
    g1[7] = 0;
    const i32x4 z4 = {0, 0, 0, 0};
    const i32x8 z8 = {0, 0, 0, 0, 0, 0, 0, 0};
    __builtin_amdgcn_tensor_load_to_lds(g0, g1, z4, z4, z8, 0);
}

// ---------------------------------------------------------------------------
// Kernel 1: qkv = x @ w_qkv.T, split to per-head fp16 [B,H,N,64] arrays.
// q is pre-scaled by 1/8 (exact), k rows are L2-normalized (shared_qk).
// Block: 128 threads (4 waves), computes a 64(M)x64(N) tile, K = 512.
// ---------------------------------------------------------------------------
__global__ __launch_bounds__(128) void k_qkv(const float* __restrict__ x,
                                             const float* __restrict__ wqkv,
                                             _Float16* __restrict__ qf,
                                             _Float16* __restrict__ kf,
                                             _Float16* __restrict__ vf) {
    __shared__ _Float16 xs[64 * 72];   // A tile (rows x K-chunk), padded stride
    __shared__ _Float16 wsb[64 * 72];  // B tile (cols x K-chunk)

    const int tid  = threadIdx.x;
    const int wv   = tid >> 5;
    const int lane = tid & 31;
    const int half = lane >> 4;
    const int n16  = lane & 15;
    const int rowbase = blockIdx.x * 64;
    const int colbase = blockIdx.y * 64;

    v8f acc[4];
#pragma unroll
    for (int t = 0; t < 4; ++t) acc[t] = vzero8f();

    for (int kc = 0; kc < 8; ++kc) {  // K chunks of 64
#pragma unroll
        for (int it = 0; it < 8; ++it) {
            const int idx = tid + it * 128;       // 0..1023
            const int r   = idx >> 4;             // 0..63
            const int d4  = (idx & 15) << 2;      // 0..60
            const float4 xv = *(const float4*)(x + (size_t)(rowbase + r) * DIMC + kc * 64 + d4);
            v4h hx; hx[0] = (_Float16)xv.x; hx[1] = (_Float16)xv.y;
                    hx[2] = (_Float16)xv.z; hx[3] = (_Float16)xv.w;
            *(v4h*)(xs + r * 72 + d4) = hx;
            const float4 wv4 = *(const float4*)(wqkv + (size_t)(colbase + r) * DIMC + kc * 64 + d4);
            v4h hw; hw[0] = (_Float16)wv4.x; hw[1] = (_Float16)wv4.y;
                    hw[2] = (_Float16)wv4.z; hw[3] = (_Float16)wv4.w;
            *(v4h*)(wsb + r * 72 + d4) = hw;
        }
        __syncthreads();
#pragma unroll
        for (int ks = 0; ks < 2; ++ks) {  // two 16x16x32 K-steps per chunk
            const _Float16* ap = xs + (16 * wv + n16) * 72 + ks * 32 + 8 * half;
            const v16h A = ldpair(ap, ap + 16);
            v16h Bf[4];
#pragma unroll
            for (int t = 0; t < 4; ++t) {
                const _Float16* bp = wsb + (16 * t + n16) * 72 + ks * 32 + 8 * half;
                Bf[t] = ldpair(bp, bp + 16);
            }
#pragma unroll
            for (int t = 0; t < 4; ++t) acc[t] = wmma_f16(A, Bf[t], acc[t]);
        }
        __syncthreads();
    }

    // Epilogue: this 64-col tile is exactly one head of one of q/k/v.
    const int seg = colbase >> 9;          // 0=q 1=k 2=v
    const int h   = (colbase >> 6) & 7;
    float invn[8];
    if (seg == 1) {
#pragma unroll
        for (int r = 0; r < 8; ++r) {
            float ss = 0.0f;
#pragma unroll
            for (int t = 0; t < 4; ++t) { const float v = acc[t][r]; ss += v * v; }
            ss += __shfl_xor(ss, 1);
            ss += __shfl_xor(ss, 2);
            ss += __shfl_xor(ss, 4);
            ss += __shfl_xor(ss, 8);   // reduce within 16-lane half (same row set)
            invn[r] = 1.0f / fmaxf(sqrtf(ss), 1e-12f);
        }
    }
    _Float16* dst = (seg == 0) ? qf : (seg == 1) ? kf : vf;
    const float mul0 = (seg == 0) ? 0.125f : 1.0f;   // fold attention scale into q
#pragma unroll
    for (int t = 0; t < 4; ++t) {
        const int d = 16 * t + n16;
#pragma unroll
        for (int r = 0; r < 8; ++r) {
            const int grow = rowbase + 16 * wv + r + 8 * half;
            const int b = grow >> 13;       // /8192
            const int n = grow & (NTOK - 1);
            float v = acc[t][r] * mul0;
            if (seg == 1) v *= invn[r];
            dst[(((size_t)(b * NHEADS + h)) * NTOK + n) * HD + d] = (_Float16)v;
        }
    }
}

// ---------------------------------------------------------------------------
// Kernel 2: sliding-window causal attention, one block per (b,h,window).
// 256 threads (8 waves); each wave owns a 16-row strip of the 128x256 logits.
// q/k tiles staged by the Tensor Data Mover; v transposed through VGPRs.
// ---------------------------------------------------------------------------
__global__ __launch_bounds__(256) void k_attn(const _Float16* __restrict__ qf,
                                              const _Float16* __restrict__ kf,
                                              const _Float16* __restrict__ vf,
                                              _Float16* __restrict__ aout) {
    __shared__ _Float16 qs[128 * 72];        // 18 KB   q rows [128][64] (pad 72)
    __shared__ _Float16 ks[256 * 72];        // 36 KB   k rows [256][64]
    __shared__ _Float16 vt[64 * 264];        // 33 KB   v transposed [64][256] (pad 264)
    __shared__ _Float16 ps[8 * 16 * 264];    // 66 KB   probs, per-wave [16][256]

    const int tid  = threadIdx.x;
    const int wv   = tid >> 5;
    const int lane = tid & 31;
    const int half = lane >> 4;
    const int n16  = lane & 15;

    const int blk = blockIdx.x;              // ((b*H+h)*NWIN + wi)
    const int wi  = blk & (NWIN - 1);
    const int bh  = blk >> 6;
    const int h   = bh & 7;
    const int b   = bh >> 3;
    const size_t base = (size_t)bh * NTOK * HD;
    const int qrow0 = wi * WIN;

    // --- TDM staging: q window (16KB) by wave 0, k valid region by wave 1 ---
    if (wv == 0) {
        tdm_load_1d_f16((unsigned)(uintptr_t)qs,
                        qf + base + (size_t)qrow0 * HD, WIN * HD);
    } else if (wv == 1) {
        if (wi > 0) {
            tdm_load_1d_f16((unsigned)(uintptr_t)ks,
                            kf + base + (size_t)(qrow0 - WIN) * HD, 2 * WIN * HD);
        } else {
            tdm_load_1d_f16((unsigned)(uintptr_t)ks + (unsigned)(WIN * 72 * 2),
                            kf + base, WIN * HD);
        }
    }
    // Window 0: zero the padded (previous-window) half of k.
    if (wi == 0) {
        v8h zz;
#pragma unroll
        for (int i = 0; i < 8; ++i) zz[i] = (_Float16)0.0f;
#pragma unroll
        for (int it = 0; it < 4; ++it) {
            const int idx = tid + it * 256;  // 0..1023 -> rows 0..127
            const int j = idx >> 3, d8 = (idx & 7) * 8;
            *(v8h*)(ks + j * 72 + d8) = zz;
        }
    }
    // v^T staged through VGPRs (transpose).
#pragma unroll
    for (int it = 0; it < 16; ++it) {
        const int c = tid + it * 256;        // 0..4095
        const int j = c >> 4, d4 = (c & 15) * 4;
        const int j0 = qrow0 - WIN + j;
        v4h val;
#pragma unroll
        for (int i = 0; i < 4; ++i) val[i] = (_Float16)0.0f;
        if (j0 >= 0) val = *(const v4h*)(vf + base + (size_t)j0 * HD + d4);
        vt[(d4 + 0) * 264 + j] = val[0];
        vt[(d4 + 1) * 264 + j] = val[1];
        vt[(d4 + 2) * 264 + j] = val[2];
        vt[(d4 + 3) * 264 + j] = val[3];
    }
    __builtin_amdgcn_s_wait_tensorcnt(0);
    __syncthreads();

    // sim = q_scaled @ k_norm^T : 16 N-tiles, K=64 (2 WMMA k-steps)
    v8f acc[16];
#pragma unroll
    for (int t = 0; t < 16; ++t) acc[t] = vzero8f();
#pragma unroll
    for (int ksi = 0; ksi < 2; ++ksi) {
        const _Float16* ap = qs + (16 * wv + n16) * 72 + ksi * 32 + 8 * half;
        const v16h A = ldpair(ap, ap + 16);
#pragma unroll
        for (int tg = 0; tg < 4; ++tg) {
            v16h Bf[4];
#pragma unroll
            for (int u = 0; u < 4; ++u) {
                const _Float16* bp = ks + (16 * (4 * tg + u) + n16) * 72 + ksi * 32 + 8 * half;
                Bf[u] = ldpair(bp, bp + 16);
            }
#pragma unroll
            for (int u = 0; u < 4; ++u)
                acc[4 * tg + u] = wmma_f16(A, Bf[u], acc[4 * tg + u]);
        }
    }

    // Masks (pad / causal / self) applied in-register using the C-tile layout.
    const int i_lo = qrow0 + 16 * wv;
    const float NEG = -3.0e38f;
#pragma unroll
    for (int t = 0; t < 16; ++t) {
        const int jg = qrow0 - WIN + 16 * t + n16;
#pragma unroll
        for (int r = 0; r < 8; ++r) {
            const int ig = i_lo + r + 8 * half;
            float s = acc[t][r];
            if (jg > ig || jg < 0)  s = NEG;        // causal | pad
            else if (jg == ig)      s = -5.0e4f;    // shared_qk self mask
            acc[t][r] = s;
        }
    }

    // Row max via shfl reductions inside each 16-lane half.
    float rmax[8];
#pragma unroll
    for (int r = 0; r < 8; ++r) {
        float m = NEG;
#pragma unroll
        for (int t = 0; t < 16; ++t) m = fmaxf(m, acc[t][r]);
        m = fmaxf(m, __shfl_xor(m, 1));
        m = fmaxf(m, __shfl_xor(m, 2));
        m = fmaxf(m, __shfl_xor(m, 4));
        m = fmaxf(m, __shfl_xor(m, 8));
        rmax[r] = m;
    }

    // Fused: write unnormalized exp() to LDS while accumulating the row sums.
    // (1/sum is applied to the 64-wide output rows instead of the 256-wide P.)
    _Float16* pw = ps + wv * 16 * 264;
    float se[8];
#pragma unroll
    for (int r = 0; r < 8; ++r) se[r] = 0.0f;
#pragma unroll
    for (int t = 0; t < 16; ++t) {
#pragma unroll
        for (int r = 0; r < 8; ++r) {
            const int row = r + 8 * half;
            const float e = __expf(acc[t][r] - rmax[r]);
            se[r] += e;
            pw[row * 264 + 16 * t + n16] = (_Float16)e;
        }
    }
    float rinv[8];
#pragma unroll
    for (int r = 0; r < 8; ++r) {
        float s = se[r];
        s += __shfl_xor(s, 1);
        s += __shfl_xor(s, 2);
        s += __shfl_xor(s, 4);
        s += __shfl_xor(s, 8);
        rinv[r] = 1.0f / s;      // self entry guarantees s > 0
    }
    __syncthreads();

    // out = P @ V : K=256 (8 k-steps), 4 N-tiles of 16 head dims.
    v8f acc2[4];
#pragma unroll
    for (int t = 0; t < 4; ++t) acc2[t] = vzero8f();
#pragma unroll
    for (int s = 0; s < 8; ++s) {
        const _Float16* ap = pw + n16 * 264 + s * 32 + 8 * half;
        const v16h A = ldpair(ap, ap + 16);
        v16h Bf[4];
#pragma unroll
        for (int t2 = 0; t2 < 4; ++t2) {
            const _Float16* bp = vt + (16 * t2 + n16) * 264 + s * 32 + 8 * half;
            Bf[t2] = ldpair(bp, bp + 16);
        }
#pragma unroll
        for (int t2 = 0; t2 < 4; ++t2) acc2[t2] = wmma_f16(A, Bf[t2], acc2[t2]);
    }
    // Write to head-interleaved [B, N, 512] fp16 (ready for the proj GEMM).
#pragma unroll
    for (int t2 = 0; t2 < 4; ++t2) {
        const int d = 16 * t2 + n16;
#pragma unroll
        for (int r = 0; r < 8; ++r) {
            const int n = qrow0 + 16 * wv + r + 8 * half;
            aout[((size_t)b * NTOK + n) * DIMC + h * HD + d] =
                (_Float16)(acc2[t2][r] * rinv[r]);
        }
    }
}

// ---------------------------------------------------------------------------
// Kernel 3: out = attn @ w_proj.T + b_proj  (fp16 A, fp16-converted W, fp32 out)
// ---------------------------------------------------------------------------
__global__ __launch_bounds__(128) void k_proj(const _Float16* __restrict__ a,
                                              const float* __restrict__ wp,
                                              const float* __restrict__ bias,
                                              float* __restrict__ out) {
    __shared__ _Float16 as[64 * 72];
    __shared__ _Float16 wsb[64 * 72];

    const int tid  = threadIdx.x;
    const int wv   = tid >> 5;
    const int lane = tid & 31;
    const int half = lane >> 4;
    const int n16  = lane & 15;
    const int rowbase = blockIdx.x * 64;
    const int colbase = blockIdx.y * 64;

    v8f acc[4];
#pragma unroll
    for (int t = 0; t < 4; ++t) acc[t] = vzero8f();

    for (int kc = 0; kc < 8; ++kc) {
#pragma unroll
        for (int it = 0; it < 4; ++it) {     // A: 512 v8h chunks
            const int idx = tid + it * 128;
            const int r = idx >> 3, d8 = (idx & 7) * 8;
            *(v8h*)(as + r * 72 + d8) = *(const v8h*)(a + (size_t)(rowbase + r) * DIMC + kc * 64 + d8);
        }
#pragma unroll
        for (int it = 0; it < 8; ++it) {     // W: 1024 float4 chunks, cvt fp16
            const int idx = tid + it * 128;
            const int r = idx >> 4, d4 = (idx & 15) << 2;
            const float4 wv4 = *(const float4*)(wp + (size_t)(colbase + r) * DIMC + kc * 64 + d4);
            v4h hw; hw[0] = (_Float16)wv4.x; hw[1] = (_Float16)wv4.y;
                    hw[2] = (_Float16)wv4.z; hw[3] = (_Float16)wv4.w;
            *(v4h*)(wsb + r * 72 + d4) = hw;
        }
        __syncthreads();
#pragma unroll
        for (int ks = 0; ks < 2; ++ks) {
            const _Float16* ap = as + (16 * wv + n16) * 72 + ks * 32 + 8 * half;
            const v16h A = ldpair(ap, ap + 16);
            v16h Bf[4];
#pragma unroll
            for (int t = 0; t < 4; ++t) {
                const _Float16* bp = wsb + (16 * t + n16) * 72 + ks * 32 + 8 * half;
                Bf[t] = ldpair(bp, bp + 16);
            }
#pragma unroll
            for (int t = 0; t < 4; ++t) acc[t] = wmma_f16(A, Bf[t], acc[t]);
        }
        __syncthreads();
    }
#pragma unroll
    for (int t = 0; t < 4; ++t) {
        const int c = colbase + 16 * t + n16;
        const float bb = bias[c];
#pragma unroll
        for (int r = 0; r < 8; ++r) {
            const int grow = rowbase + 16 * wv + r + 8 * half;
            out[(size_t)grow * DIMC + c] = acc[t][r] + bb;
        }
    }
}

extern "C" void kernel_launch(void* const* d_in, const int* in_sizes, int n_in,
                              void* d_out, int out_size, void* d_ws, size_t ws_size,
                              hipStream_t stream) {
    const float* x     = (const float*)d_in[0];   // [4,8192,512]
    const float* wqkv  = (const float*)d_in[1];   // [1536,512]
    const float* wproj = (const float*)d_in[2];   // [512,512]
    const float* bproj = (const float*)d_in[3];   // [512]
    float* out = (float*)d_out;                   // [4,8192,512]

    _Float16* qf = (_Float16*)d_ws;               // [B,H,N,64] pre-scaled q
    _Float16* kf = qf + HEADSZ;                   // [B,H,N,64] L2-normalized k
    _Float16* vf = kf + HEADSZ;                   // [B,H,N,64]
    _Float16* at = vf + HEADSZ;                   // [B,N,512] attention output

    k_qkv <<<dim3(512, 24), 128, 0, stream>>>(x, wqkv, qf, kf, vf);
    k_attn<<<dim3(BATCH * NHEADS * NWIN), 256, 0, stream>>>(qf, kf, vf, at);
    k_proj<<<dim3(512, 8), 128, 0, stream>>>(at, wproj, bproj, out);
}
